// ImmuneFractalBlock_80994493268354
// MI455X (gfx1250) — matrix operand order repaired and checked by reading, not compiled, not run on hardware
//
#include <hip/hip_runtime.h>

// ---------------------------------------------------------------------------
// Types / helpers
// ---------------------------------------------------------------------------
typedef __bf16 bf16_t;
typedef bf16_t v8bf  __attribute__((ext_vector_type(8)));
typedef bf16_t v16bf __attribute__((ext_vector_type(16)));
typedef float  v8f   __attribute__((ext_vector_type(8)));

#define DEV __device__ __forceinline__

DEV bf16_t f2bf(float f) {
  unsigned u = __builtin_bit_cast(unsigned, f);
  unsigned r = (u + 0x7fffu + ((u >> 16) & 1u)) >> 16;
  return __builtin_bit_cast(bf16_t, (unsigned short)r);
}
DEV float bf2f(bf16_t b) {
  unsigned u = ((unsigned)__builtin_bit_cast(unsigned short, b)) << 16;
  return __builtin_bit_cast(float, u);
}

// A-operand fragment (16x32 bf16, M rows striped in-lane):
// lanes 0-15: row=lrow, K = k0+{0..7, 16..23}; lanes 16-31: K = k0+{8..15, 24..31}
DEV v16bf load_a_frag(const bf16_t* base, int stride, int row, int k0, int hi) {
  const bf16_t* p = base + (size_t)row * stride + k0 + hi * 8;
  v8bf lo = *(const v8bf*)p;
  v8bf hh = *(const v8bf*)(p + 16);
  return __builtin_shufflevector(lo, hh, 0,1,2,3,4,5,6,7,8,9,10,11,12,13,14,15);
}
// B-operand fragment (32x16 bf16, stored as [N,K] row-major):
// lanes 0-15: col N=row, K = k0+{0..7, 16..23} ; lanes 16-31: K = k0+{8..15, 24..31}
DEV v16bf load_b_frag(const bf16_t* base, int stride, int row, int k0, int hi) {
  const bf16_t* p = base + (size_t)row * stride + k0 + hi * 8;
  v8bf lo = *(const v8bf*)p;
  v8bf hh = *(const v8bf*)(p + 16);
  return __builtin_shufflevector(lo, hh, 0,1,2,3,4,5,6,7,8,9,10,11,12,13,14,15);
}
DEV v8f wmma_bf16(v16bf a, v16bf b, v8f c) {
  return __builtin_amdgcn_wmma_f32_16x16x32_bf16(false, a, false, b, (short)0, c,
                                                 false, false);
}
DEV float redmax16(float v) {
#pragma unroll
  for (int m = 1; m <= 8; m <<= 1) v = fmaxf(v, __shfl_xor(v, m, 32));
  return v;
}
DEV float redsum16(float v) {
#pragma unroll
  for (int m = 1; m <= 8; m <<= 1) v += __shfl_xor(v, m, 32);
  return v;
}

// CDNA5 async DMA: global -> LDS, no VGPR round trip, tracked by ASYNCcnt.
DEV void async_ld_b128(unsigned lds_off, const void* gaddr) {
  asm volatile("global_load_async_to_lds_b128 %0, %1, off"
               :
               : "v"(lds_off), "v"((unsigned long long)gaddr)
               : "memory");
}
DEV void wait_asynccnt0() {
  asm volatile("s_wait_asynccnt 0x0" ::: "memory");
}
DEV unsigned lds_off_of(const void* p) {
  // generic LDS address: low 32 bits == LDS byte offset (ISA 10.2 aperture map)
  return (unsigned)(unsigned long long)p;
}

// ---------------------------------------------------------------------------
// 1) Immune weight assembly: W[o, :] = rs[o] * concat(sm(vl)@vb, sm(dl)@db,
//    sm(jl)@jb), emitted as bf16 row-major [out_f, in_f]  (== WMMA B layout)
// ---------------------------------------------------------------------------
DEV void softmax_range(const float* l, float* p, int a, int b) {
  float mx = -3.0e38f;
  for (int i = a; i < b; ++i) mx = fmaxf(mx, l[i]);
  float s = 0.f;
  for (int i = a; i < b; ++i) { float e = __expf(l[i] - mx); p[i] = e; s += e; }
  float inv = 1.f / s;
  for (int i = a; i < b; ++i) p[i] *= inv;
}

__global__ __launch_bounds__(128) void assemble_w_kernel(
    const float* __restrict__ vb, const float* __restrict__ db,
    const float* __restrict__ jb, const float* __restrict__ vl,
    const float* __restrict__ dl, const float* __restrict__ jl,
    const float* __restrict__ rs, bf16_t* __restrict__ W,
    int in_f, int seg, int jdim) {
  __shared__ float slog[112];
  __shared__ float sp[112];
  const int o = blockIdx.x;
  const int tid = threadIdx.x;
  if (tid < 64)       slog[tid] = vl[o * 64 + tid];
  else if (tid < 96)  slog[tid] = dl[o * 32 + (tid - 64)];
  else if (tid < 112) slog[tid] = jl[o * 16 + (tid - 96)];
  __syncthreads();
  if (tid == 0)  softmax_range(slog, sp, 0, 64);
  if (tid == 32) softmax_range(slog, sp, 64, 96);
  if (tid == 64) softmax_range(slog, sp, 96, 112);
  __syncthreads();
  const float sc = rs[o];
  for (int k = tid; k < in_f; k += 128) {
    float acc = 0.f;
    if (k < seg) {
      for (int g = 0; g < 64; ++g) acc += sp[g] * vb[(size_t)g * seg + k];
    } else if (k < 2 * seg) {
      const int kk = k - seg;
      for (int g = 0; g < 32; ++g) acc += sp[64 + g] * db[(size_t)g * seg + kk];
    } else {
      const int kk = k - 2 * seg;
      for (int g = 0; g < 16; ++g) acc += sp[96 + g] * jb[(size_t)g * jdim + kk];
    }
    W[(size_t)o * in_f + k] = f2bf(acc * sc);
  }
}

// ---------------------------------------------------------------------------
// 2) RMSNorm (fp32 in -> bf16 out), eps = FLT_EPSILON (jax finfo(f32).eps)
// ---------------------------------------------------------------------------
__global__ __launch_bounds__(256) void rmsnorm_kernel(
    const float* __restrict__ x, const float* __restrict__ w,
    bf16_t* __restrict__ out, int D) {
  __shared__ float red[8];
  __shared__ float sscale;
  const int row = blockIdx.x;
  const int tid = threadIdx.x;
  const float* xr = x + (size_t)row * D;
  float ss = 0.f;
  for (int i = tid; i < D; i += 256) { float v = xr[i]; ss += v * v; }
#pragma unroll
  for (int m = 1; m <= 16; m <<= 1) ss += __shfl_xor(ss, m, 32);
  if ((tid & 31) == 0) red[tid >> 5] = ss;
  __syncthreads();
  if (tid == 0) {
    float t = 0.f;
#pragma unroll
    for (int i = 0; i < 8; ++i) t += red[i];
    sscale = rsqrtf(t / (float)D + 1.1920929e-07f);
  }
  __syncthreads();
  const float scale = sscale;
  for (int i = tid; i < D; i += 256)
    out[(size_t)row * D + i] = f2bf(xr[i] * scale * w[i]);
}

// ---------------------------------------------------------------------------
// 3) Tiled WMMA GEMM:  C[M,N] = A[M,K](bf16) @ Bw[N,K](bf16)^T, fp32 accum.
//    Block 256 thr = 8 waves, block tile 128x128, wave tile 32x64 (2x4 frags).
//    Double-buffered async-to-LDS staging: per K-step each thread issues one
//    16B global_load_async_to_lds_b128 for A and one for B (128x32 bf16 tiles),
//    next tile's DMA overlaps current tile's WMMAs; ASYNCcnt + barrier sync.
//    Epilogues: 0=QKV scatter, 1=resid+fp32 out, 2=silu bf16, 3=mul bf16
// ---------------------------------------------------------------------------
enum { EPI_QKV = 0, EPI_RESID = 1, EPI_SILU = 2, EPI_MUL = 3 };

template <int MODE>
__global__ __launch_bounds__(256) void gemm_bf16_kernel(
    const bf16_t* __restrict__ A, const bf16_t* __restrict__ Bw, int K, int N,
    float* __restrict__ outf, const float* __restrict__ resid,
    const bf16_t* __restrict__ mulb, bf16_t* __restrict__ outb,
    bf16_t* __restrict__ Qb, bf16_t* __restrict__ Kb,
    bf16_t* __restrict__ Vtb) {
  __shared__ __align__(32) bf16_t sA[2][128 * 32];
  __shared__ __align__(32) bf16_t sB[2][128 * 32];

  const int tid = threadIdx.x;
  const int lane = tid & 31;
  const int wid  = tid >> 5;
  const int wr = wid >> 1, wc = wid & 1;
  const int lrow = lane & 15, hi = lane >> 4;
  const int mBlock = blockIdx.y * 128;
  const int nBlock = blockIdx.x * 128;
  const int mBase = wr * 32;  // within block tile
  const int nBase = wc * 64;

  // cooperative tile staging: thread -> (row, 16-elem half) of the 128x32 tile
  const int ldRow  = tid >> 1;
  const int ldHalf = (tid & 1) * 16;
  const unsigned ldsA0 = lds_off_of(&sA[0][ldRow * 32 + ldHalf]);
  const unsigned ldsB0 = lds_off_of(&sB[0][ldRow * 32 + ldHalf]);
  const unsigned stageBytes = (unsigned)(128 * 32 * sizeof(bf16_t));
  const bf16_t* gA = A + (size_t)(mBlock + ldRow) * K + ldHalf;
  const bf16_t* gB = Bw + (size_t)(nBlock + ldRow) * K + ldHalf;

  auto stage = [&](int st, int k0) {
    async_ld_b128(ldsA0 + st * stageBytes, gA + k0);
    async_ld_b128(ldsB0 + st * stageBytes, gB + k0);
  };

  stage(0, 0);
  wait_asynccnt0();
  __syncthreads();

  v8f acc[2][4] = {};
  int cur = 0;
  for (int k0 = 0; k0 < K; k0 += 32) {
    if (k0 + 32 < K) stage(cur ^ 1, k0 + 32);  // overlap DMA with WMMA

    const bf16_t* tA = &sA[cur][0];
    const bf16_t* tB = &sB[cur][0];
    v16bf a0 = load_a_frag(tA, 32, mBase + lrow, 0, hi);
    v16bf a1 = load_a_frag(tA, 32, mBase + 16 + lrow, 0, hi);
    v16bf b0 = load_b_frag(tB, 32, nBase + lrow, 0, hi);
    v16bf b1 = load_b_frag(tB, 32, nBase + 16 + lrow, 0, hi);
    v16bf b2 = load_b_frag(tB, 32, nBase + 32 + lrow, 0, hi);
    v16bf b3 = load_b_frag(tB, 32, nBase + 48 + lrow, 0, hi);
    acc[0][0] = wmma_bf16(a0, b0, acc[0][0]);
    acc[0][1] = wmma_bf16(a0, b1, acc[0][1]);
    acc[0][2] = wmma_bf16(a0, b2, acc[0][2]);
    acc[0][3] = wmma_bf16(a0, b3, acc[0][3]);
    acc[1][0] = wmma_bf16(a1, b0, acc[1][0]);
    acc[1][1] = wmma_bf16(a1, b1, acc[1][1]);
    acc[1][2] = wmma_bf16(a1, b2, acc[1][2]);
    acc[1][3] = wmma_bf16(a1, b3, acc[1][3]);

    wait_asynccnt0();   // next stage's DMA complete (LDS written)
    __syncthreads();    // all waves done reading cur, next stage visible
    cur ^= 1;
  }

#pragma unroll
  for (int i = 0; i < 2; ++i)
#pragma unroll
    for (int j = 0; j < 4; ++j)
#pragma unroll
      for (int r = 0; r < 8; ++r) {
        const int row = mBlock + mBase + i * 16 + r + hi * 8;  // token index
        const int col = nBlock + nBase + j * 16 + lrow;        // out feature
        float v = acc[i][j][r];
        if (MODE == EPI_QKV) {
          const int which = col >> 10;
          const int rn = col & 1023;
          const int h = rn >> 6, d = rn & 63;
          const int b = row >> 11, t = row & 2047;
          const size_t bh = (size_t)(b * 16 + h);
          if (which == 0)
            Qb[(bh * 2048 + t) * 64 + d] = f2bf(v * 0.125f);  // 1/sqrt(64)
          else if (which == 1)
            Kb[(bh * 2048 + t) * 64 + d] = f2bf(v);
          else
            Vtb[(bh * 64 + d) * 2048 + t] = f2bf(v);          // V transposed
        } else if (MODE == EPI_RESID) {
          const size_t idx = (size_t)row * N + col;
          outf[idx] = resid[idx] + v;
        } else if (MODE == EPI_SILU) {
          outb[(size_t)row * N + col] = f2bf(v / (1.f + __expf(-v)));
        } else {  // EPI_MUL
          const size_t idx = (size_t)row * N + col;
          outb[idx] = f2bf(v * bf2f(mulb[idx]));
        }
      }
}

// ---------------------------------------------------------------------------
// 4) Causal flash attention, bf16 WMMA. Grid: (T/64, B*H), block 128 = 4 waves,
//    each wave owns 16 q-rows; 32 keys per outer iteration (two 16-key S tiles
//    fused into one K=32 P@V WMMA). P transposed C-layout -> A-layout via LDS.
// ---------------------------------------------------------------------------
__global__ __launch_bounds__(128) void attn_kernel(
    const bf16_t* __restrict__ Qb, const bf16_t* __restrict__ Kb,
    const bf16_t* __restrict__ Vtb, bf16_t* __restrict__ Ob) {
  constexpr int T = 2048, HD = 64, H = 16;
  const int lane = threadIdx.x & 31;
  const int w = threadIdx.x >> 5;
  const int lrow = lane & 15, hi = lane >> 4;
  const int bh = blockIdx.y;
  const int b = bh >> 4, h = bh & 15;
  const bf16_t* Q  = Qb + (size_t)bh * T * HD;
  const bf16_t* Kp = Kb + (size_t)bh * T * HD;
  const bf16_t* Vt = Vtb + (size_t)bh * HD * T;
  const int qbase = blockIdx.x * 64 + w * 16;

  // Q A-fragments (hd split into two K=32 chunks)
  v16bf aq0 = load_a_frag(Q, HD, qbase + lrow, 0, hi);
  v16bf aq1 = load_a_frag(Q, HD, qbase + lrow, 32, hi);

  v8f acc[4] = {};
  float m_r[8], l_r[8];
#pragma unroll
  for (int r = 0; r < 8; ++r) { m_r[r] = -3.0e38f; l_r[r] = 0.f; }

  __shared__ __align__(32) bf16_t pbuf[4][16 * 32];
  bf16_t* pw = &pbuf[w][0];
  const v8f zf = {0.f, 0.f, 0.f, 0.f, 0.f, 0.f, 0.f, 0.f};

  const int nkb = (qbase + 16 + 31) >> 5;  // causal: only blocks <= diagonal
  for (int kb32 = 0; kb32 < nkb; ++kb32) {
    const int kb = kb32 * 32;
    // S = Q @ K^T for two 16-key subtiles (B operand from row-major K[t,hd])
    v16bf bk0a = load_b_frag(Kp, HD, kb + lrow, 0, hi);
    v16bf bk0b = load_b_frag(Kp, HD, kb + lrow, 32, hi);
    v16bf bk1a = load_b_frag(Kp, HD, kb + 16 + lrow, 0, hi);
    v16bf bk1b = load_b_frag(Kp, HD, kb + 16 + lrow, 32, hi);
    v8f s0 = wmma_bf16(aq0, bk0a, zf);
    s0 = wmma_bf16(aq1, bk0b, s0);
    v8f s1 = wmma_bf16(aq0, bk1a, zf);
    s1 = wmma_bf16(aq1, bk1b, s1);

    const int key0 = kb + lrow, key1 = kb + 16 + lrow;
#pragma unroll
    for (int r = 0; r < 8; ++r) {
      const int qrow = qbase + r + hi * 8;
      float v0 = (key0 <= qrow) ? s0[r] : -3.0e38f;
      float v1 = (key1 <= qrow) ? s1[r] : -3.0e38f;
      float mx = redmax16(fmaxf(v0, v1));
      float mnew = fmaxf(m_r[r], mx);
      float sc = __expf(m_r[r] - mnew);
      m_r[r] = mnew;
      float p0 = __expf(v0 - mnew);
      float p1 = __expf(v1 - mnew);
      float rsum = redsum16(p0 + p1);
      l_r[r] = l_r[r] * sc + rsum;
#pragma unroll
      for (int c = 0; c < 4; ++c) acc[c][r] *= sc;
      // stage P (C-layout) into LDS as row-major 16x32 bf16
      pw[(r + hi * 8) * 32 + lrow] = f2bf(p0);
      pw[(r + hi * 8) * 32 + 16 + lrow] = f2bf(p1);
    }
    __builtin_amdgcn_wave_barrier();
    // reload P as an A-operand fragment (16x32, K = 32 keys)
    v16bf pa = load_a_frag(pw, 32, lrow, 0, hi);
    __builtin_amdgcn_wave_barrier();
    // acc += P @ V  (B operand from V^T[hd, t], N = hd chunk, K = keys)
#pragma unroll
    for (int c = 0; c < 4; ++c) {
      v16bf bv = load_b_frag(Vt, T, c * 16 + lrow, kb, hi);
      acc[c] = wmma_bf16(pa, bv, acc[c]);
    }
  }

  float inv[8];
#pragma unroll
  for (int r = 0; r < 8; ++r) inv[r] = 1.f / l_r[r];
#pragma unroll
  for (int c = 0; c < 4; ++c)
#pragma unroll
    for (int r = 0; r < 8; ++r) {
      const int t = qbase + r + hi * 8;
      Ob[((size_t)(b * T + t)) * (H * HD) + h * HD + c * 16 + lrow] =
          f2bf(acc[c][r] * inv[r]);
    }
}

// ---------------------------------------------------------------------------
// Host: launch pipeline
// ---------------------------------------------------------------------------
extern "C" void kernel_launch(void* const* d_in, const int* in_sizes, int n_in,
                              void* d_out, int out_size, void* d_ws,
                              size_t ws_size, hipStream_t stream) {
  (void)in_sizes; (void)n_in; (void)out_size; (void)ws_size;
  constexpr int Dm = 1024, FF = 2048, T = 2048, Bt = 2;
  constexpr int M = Bt * T;  // 4096 tokens

  const float* x = (const float*)d_in[0];
  // immune params: vb, db, jb, vl, dl, jl, rs
  const float* const* qkv = (const float* const*)&d_in[1];
  const float* const* op  = (const float* const*)&d_in[8];
  const float* const* gp  = (const float* const*)&d_in[15];
  const float* const* up  = (const float* const*)&d_in[22];
  const float* const* dp  = (const float* const*)&d_in[29];
  const float* norm1_w = (const float*)d_in[36];
  const float* norm2_w = (const float*)d_in[37];
  float* out = (float*)d_out;

  // Workspace carve (aligned 256B). ~76 MB total with reuse.
  char* base = (char*)d_ws;
  size_t off = 0;
  auto carve = [&](size_t bytes) {
    char* r = base + off;
    off += (bytes + 255) & ~(size_t)255;
    return r;
  };
  bf16_t* Wqkv = (bf16_t*)carve((size_t)3 * Dm * Dm * 2);
  bf16_t* Wo   = (bf16_t*)carve((size_t)Dm * Dm * 2);
  bf16_t* Wg   = (bf16_t*)carve((size_t)FF * Dm * 2);
  bf16_t* Wu   = (bf16_t*)carve((size_t)FF * Dm * 2);
  bf16_t* Wd   = (bf16_t*)carve((size_t)Dm * FF * 2);
  bf16_t* R5   = (bf16_t*)carve((size_t)M * FF * 2);  // h1 | Obf ; later siluG
  bf16_t* R6   = (bf16_t*)carve((size_t)M * FF * 2);  // Qb | Kb  ; later ff
  bf16_t* R7   = (bf16_t*)carve((size_t)M * Dm * 2);  // Vt ; later h2
  float*  x1   = (float*)carve((size_t)M * Dm * 4);

  bf16_t* h1 = R5;
  bf16_t* Obf = R5 + (size_t)M * Dm;
  bf16_t* siluG = R5;
  bf16_t* Qbuf = R6;
  bf16_t* Kbuf = R6 + (size_t)M * Dm;
  bf16_t* ff = R6;
  bf16_t* Vtb = R7;
  bf16_t* h2 = R7;

  // 1) assemble weights (bf16 [out,in])
  assemble_w_kernel<<<dim3(3 * Dm), dim3(128), 0, stream>>>(
      qkv[0], qkv[1], qkv[2], qkv[3], qkv[4], qkv[5], qkv[6], Wqkv, Dm, 341, 342);
  assemble_w_kernel<<<dim3(Dm), dim3(128), 0, stream>>>(
      op[0], op[1], op[2], op[3], op[4], op[5], op[6], Wo, Dm, 341, 342);
  assemble_w_kernel<<<dim3(FF), dim3(128), 0, stream>>>(
      gp[0], gp[1], gp[2], gp[3], gp[4], gp[5], gp[6], Wg, Dm, 341, 342);
  assemble_w_kernel<<<dim3(FF), dim3(128), 0, stream>>>(
      up[0], up[1], up[2], up[3], up[4], up[5], up[6], Wu, Dm, 341, 342);
  assemble_w_kernel<<<dim3(Dm), dim3(128), 0, stream>>>(
      dp[0], dp[1], dp[2], dp[3], dp[4], dp[5], dp[6], Wd, FF, 682, 684);

  // 2) h1 = rmsnorm(x)
  rmsnorm_kernel<<<dim3(M), dim3(256), 0, stream>>>(x, norm1_w, h1, Dm);

  // 3) qkv = h1 @ Wqkv^T, scattered to Q(scaled)/K/V^T head-major bf16
  gemm_bf16_kernel<EPI_QKV><<<dim3((3 * Dm) / 128, M / 128), dim3(256), 0, stream>>>(
      h1, Wqkv, Dm, 3 * Dm, nullptr, nullptr, nullptr, nullptr, Qbuf, Kbuf, Vtb);

  // 4) attention
  attn_kernel<<<dim3(T / 64, Bt * 16), dim3(128), 0, stream>>>(Qbuf, Kbuf, Vtb, Obf);

  // 5) x1 = x + attn_out @ Wo^T
  gemm_bf16_kernel<EPI_RESID><<<dim3(Dm / 128, M / 128), dim3(256), 0, stream>>>(
      Obf, Wo, Dm, Dm, x1, x, nullptr, nullptr, nullptr, nullptr, nullptr);

  // 6) h2 = rmsnorm(x1)
  rmsnorm_kernel<<<dim3(M), dim3(256), 0, stream>>>(x1, norm2_w, h2, Dm);

  // 7) siluG = silu(h2 @ Wg^T)
  gemm_bf16_kernel<EPI_SILU><<<dim3(FF / 128, M / 128), dim3(256), 0, stream>>>(
      h2, Wg, Dm, FF, nullptr, nullptr, nullptr, siluG, nullptr, nullptr, nullptr);

  // 8) ff = siluG * (h2 @ Wu^T)
  gemm_bf16_kernel<EPI_MUL><<<dim3(FF / 128, M / 128), dim3(256), 0, stream>>>(
      h2, Wu, Dm, FF, nullptr, nullptr, siluG, ff, nullptr, nullptr, nullptr);

  // 9) out = x1 + ff @ Wd^T
  gemm_bf16_kernel<EPI_RESID><<<dim3(Dm / 128, M / 128), dim3(256), 0, stream>>>(
      ff, Wd, FF, Dm, out, x1, nullptr, nullptr, nullptr, nullptr, nullptr);
}
